// plpLayer_87016037417454
// MI455X (gfx1250) — compile-verified, matrix operand order checked by reading, not run on hardware
//
#include <hip/hip_runtime.h>
#include <hip/hip_bf16.h>
#include <math.h>

#define N_NODES 50000
#define N_EDGES 1600000
#define N_FEAT  256
#define HIDDEN  32
#define N_CLASS 64

typedef __attribute__((ext_vector_type(16))) __bf16 v16bf;
typedef __attribute__((ext_vector_type(8)))  float  v8f;

// ---------------------------------------------------------------------------
// Workspace layout
//   W1s : 256x32 bf16, pre-swizzled into wave32 B-fragment order (16 KB)
//   W2s : 32x64  bf16, pre-swizzled                               (4 KB)
//   PA  : p ping buffer  [N_NODES,64] f32 (12.8 MB, L2-resident)
//   PB  : accumulator    [N_NODES,64] f32 (12.8 MB, L2-resident)
// ---------------------------------------------------------------------------
#define OFF_W1S 0
#define OFF_W2S 16384
#define OFF_PA  32768
#define OFF_PB  (32768 + (size_t)N_NODES * N_CLASS * 4)

// =============================================================
// prep: zero accumulator; swizzle W1/W2 (f32) into bf16 fragment
// layout: frag[kblock][tile][lane][16], lane group koff = (lane>>4)*8,
// vec element j -> K = kb*32 + koff + (j<8 ? j : j+8)  (ISA 7.12.2)
// =============================================================
__global__ __launch_bounds__(256) void prep_kernel(
    const float* __restrict__ W1, const float* __restrict__ W2,
    __bf16* __restrict__ W1s, __bf16* __restrict__ W2s,
    float4* __restrict__ B4) {
  int idx = blockIdx.x * 256 + threadIdx.x;
  if (idx < (N_NODES * N_CLASS / 4)) {
    B4[idx] = make_float4(0.f, 0.f, 0.f, 0.f);
  }
  if (idx < 8192) {  // W1s: 8 kblocks * 2 tiles * 32 lanes * 16
    int j    = idx & 15;
    int lane = (idx >> 4) & 31;
    int tile = (idx >> 9) & 1;
    int kb   = idx >> 10;
    int koff = (lane >> 4) * 8;
    int jj   = (j < 8) ? j : (j + 8);
    int K    = kb * 32 + koff + jj;
    int n    = (lane & 15) + 16 * tile;
    W1s[idx] = (__bf16)W1[K * HIDDEN + n];
  }
  if (idx < 2048) {  // W2s: 4 tiles * 32 lanes * 16 (single K=32 block)
    int j    = idx & 15;
    int lane = (idx >> 4) & 31;
    int tile = idx >> 9;
    int koff = (lane >> 4) * 8;
    int jj   = (j < 8) ? j : (j + 8);
    int K    = koff + jj;
    int n    = 16 * tile + (lane & 15);
    W2s[idx] = (__bf16)W2[K * N_CLASS + n];
  }
}

// pack two contiguous 8-float runs into a wave32 bf16 A/B fragment
__device__ inline v16bf pack_frag(const float* __restrict__ lo,
                                  const float* __restrict__ hi) {
  v16bf r;
#pragma unroll
  for (int j = 0; j < 8; ++j) {
    r[j]     = (__bf16)lo[j];
    r[j + 8] = (__bf16)hi[j];
  }
  return r;
}

// =============================================================
// Fused MLP + softmax: 1 wave per 16-row tile.
//   h = relu(x@W1 + b1)   -> 16 x v_wmma_f32_16x16x32_bf16
//   p = softmax(h@W2 + b2) -> 4 x v_wmma_f32_16x16x32_bf16
// =============================================================
__global__ __launch_bounds__(32) void mlp_softmax_kernel(
    const float* __restrict__ x,
    const float* __restrict__ bias1, const float* __restrict__ bias2,
    const __bf16* __restrict__ W1s, const __bf16* __restrict__ W2s,
    float* __restrict__ P) {
  __shared__ float hs[16 * HIDDEN];   // h tile, row-major
  __shared__ float ps[16 * N_CLASS];  // logits tile, row-major

  const int lane    = threadIdx.x;          // wave32
  const int rowBase = blockIdx.x * 16;
  const int r       = lane & 15;
  const int koff    = (lane >> 4) * 8;
  const int rgrp    = (lane >> 4) * 8;      // C-layout row group

  v8f c0 = {}, c1 = {};
#pragma unroll
  for (int kb = 0; kb < 8; ++kb) {
    const float* xr = x + (size_t)(rowBase + r) * N_FEAT + kb * 32 + koff;
    // A fragment: two contiguous 8-float runs (K+0..7, K+16..23)
    v16bf a = pack_frag(xr, xr + 16);
    v16bf b0 = *(const v16bf*)(W1s + ((size_t)(kb * 2 + 0) * 32 + lane) * 16);
    v16bf b1 = *(const v16bf*)(W1s + ((size_t)(kb * 2 + 1) * 32 + lane) * 16);
    c0 = __builtin_amdgcn_wmma_f32_16x16x32_bf16(false, a, false, b0,
                                                 (short)0, c0, false, false);
    c1 = __builtin_amdgcn_wmma_f32_16x16x32_bf16(false, a, false, b1,
                                                 (short)0, c1, false, false);
  }

  // bias + ReLU, C-layout -> row-major LDS
  const float bv0 = bias1[(lane & 15)];
  const float bv1 = bias1[16 + (lane & 15)];
#pragma unroll
  for (int v = 0; v < 8; ++v) {
    int row = v + rgrp;
    hs[row * HIDDEN + (lane & 15)]      = fmaxf(c0[v] + bv0, 0.f);
    hs[row * HIDDEN + 16 + (lane & 15)] = fmaxf(c1[v] + bv1, 0.f);
  }
  __syncthreads();

  // h as A fragment (K=32): runs hs[r][koff..koff+7], hs[r][koff+16..koff+23]
  v16bf ha = pack_frag(&hs[r * HIDDEN + koff], &hs[r * HIDDEN + koff + 16]);

  v8f d[4];
#pragma unroll
  for (int t = 0; t < 4; ++t) {
    float bv = bias2[t * 16 + (lane & 15)];
    v8f cc = {bv, bv, bv, bv, bv, bv, bv, bv};
    v16bf bb = *(const v16bf*)(W2s + ((size_t)(t * 32 + lane)) * 16);
    d[t] = __builtin_amdgcn_wmma_f32_16x16x32_bf16(false, ha, false, bb,
                                                   (short)0, cc, false, false);
  }
#pragma unroll
  for (int t = 0; t < 4; ++t)
#pragma unroll
    for (int v = 0; v < 8; ++v)
      ps[(v + rgrp) * N_CLASS + t * 16 + (lane & 15)] = d[t][v];
  __syncthreads();

  // row softmax: lanes 0..15 each own one row (64 elements, LDS-resident)
  if (lane < 16) {
    float* pr = &ps[lane * N_CLASS];
    float m = -3.402823466e38f;
    for (int c = 0; c < N_CLASS; ++c) m = fmaxf(m, pr[c]);
    float s = 0.f;
    for (int c = 0; c < N_CLASS; ++c) { float e = __expf(pr[c] - m); pr[c] = e; s += e; }
    float inv = 1.f / s;
    float* po = P + (size_t)(rowBase + lane) * N_CLASS;
    for (int c = 0; c < N_CLASS; ++c) po[c] = pr[c] * inv;
  }
}

// =============================================================
// SpMM scatter: one wave handles 32 edges. Lanes load edge triples
// coalesced, broadcast via shuffle; per edge all 32 lanes gather a
// float2 (256 B/edge, coalesced, L2-resident) and scatter with two
// global_atomic_add_f32. Edge stream prefetched ahead.
// =============================================================
__global__ __launch_bounds__(256) void spmm_scatter_kernel(
    const int* __restrict__ esrc, const int* __restrict__ edst,
    const float* __restrict__ eval,
    const float* __restrict__ P, float* __restrict__ B) {
  const int lane  = threadIdx.x & 31;
  const int wave  = blockIdx.x * (blockDim.x >> 5) + (threadIdx.x >> 5);
  const int eBase = wave * 32;
  if (eBase >= N_EDGES) return;

  const int e = eBase + lane;
  int s = 0, dd = 0;
  float w = 0.f;
  if (e < N_EDGES) { s = esrc[e]; dd = edst[e]; w = eval[e]; }

  // stream the edge lists ahead of use (speculative -> safe at the tail)
  __builtin_prefetch(esrc + eBase + 8192, 0, 0);
  __builtin_prefetch(edst + eBase + 8192, 0, 0);
  __builtin_prefetch(eval + eBase + 8192, 0, 0);

  const int nvalid = (N_EDGES - eBase) < 32 ? (N_EDGES - eBase) : 32;
  const int c2 = lane * 2;
  for (int i = 0; i < nvalid; ++i) {
    int   si = __shfl(s, i, 32);
    int   di = __shfl(dd, i, 32);
    float wi = __shfl(w, i, 32);
    float2 pv = *(const float2*)(P + (size_t)si * N_CLASS + c2);
    atomicAdd(B + (size_t)di * N_CLASS + c2,     pv.x * wi);
    atomicAdd(B + (size_t)di * N_CLASS + c2 + 1, pv.y * wi);
  }
}

// =============================================================
// p = 0.5*tanh(accum); accum = 0 for next round (fused re-zero)
// =============================================================
__global__ __launch_bounds__(256) void tanh_rezero_kernel(
    float4* __restrict__ A, float4* __restrict__ B, int n4) {
  int i = blockIdx.x * blockDim.x + threadIdx.x;
  if (i >= n4) return;
  float4 v = B[i];
  float4 o;
  o.x = 0.5f * tanhf(v.x);
  o.y = 0.5f * tanhf(v.y);
  o.z = 0.5f * tanhf(v.z);
  o.w = 0.5f * tanhf(v.w);
  A[i] = o;
  B[i] = make_float4(0.f, 0.f, 0.f, 0.f);
}

// =============================================================
// Final row softmax: one wave per row, float2 per lane, wave32
// shuffle reductions.
// =============================================================
__global__ __launch_bounds__(256) void softmax_out_kernel(
    const float* __restrict__ P, float* __restrict__ out) {
  const int lane = threadIdx.x & 31;
  const int row  = blockIdx.x * (blockDim.x >> 5) + (threadIdx.x >> 5);
  if (row >= N_NODES) return;
  const float2 v = *(const float2*)(P + (size_t)row * N_CLASS + lane * 2);
  float m = fmaxf(v.x, v.y);
#pragma unroll
  for (int off = 16; off; off >>= 1) m = fmaxf(m, __shfl_xor(m, off, 32));
  float e0 = __expf(v.x - m);
  float e1 = __expf(v.y - m);
  float s = e0 + e1;
#pragma unroll
  for (int off = 16; off; off >>= 1) s += __shfl_xor(s, off, 32);
  float inv = 1.f / s;
  float2 o; o.x = e0 * inv; o.y = e1 * inv;
  *(float2*)(out + (size_t)row * N_CLASS + lane * 2) = o;
}

extern "C" void kernel_launch(void* const* d_in, const int* in_sizes, int n_in,
                              void* d_out, int out_size, void* d_ws, size_t ws_size,
                              hipStream_t stream) {
  const float* x        = (const float*)d_in[0];
  const int*   edge_src = (const int*)  d_in[1];
  const int*   edge_dst = (const int*)  d_in[2];
  const float* edge_val = (const float*)d_in[3];
  const float* W1       = (const float*)d_in[4];
  const float* bias1    = (const float*)d_in[5];
  const float* W2       = (const float*)d_in[6];
  const float* bias2    = (const float*)d_in[7];

  char* ws = (char*)d_ws;
  __bf16* W1s = (__bf16*)(ws + OFF_W1S);
  __bf16* W2s = (__bf16*)(ws + OFF_W2S);
  float*  PA  = (float*)(ws + OFF_PA);
  float*  PB  = (float*)(ws + OFF_PB);

  const int n4 = N_NODES * N_CLASS / 4;  // 800000 float4s

  // zero accumulator + bf16 weight swizzle
  prep_kernel<<<(n4 + 255) / 256, 256, 0, stream>>>(W1, W2, W1s, W2s,
                                                    (float4*)PB);
  // fused MLP + softmax -> PA
  mlp_softmax_kernel<<<N_NODES / 16, 32, 0, stream>>>(x, bias1, bias2,
                                                      W1s, W2s, PA);
  // 4 propagation rounds
  const int spmmBlocks = (N_EDGES + 255) / 256;  // 8 waves/block, 32 edges/wave
  for (int it = 0; it < 4; ++it) {
    spmm_scatter_kernel<<<spmmBlocks, 256, 0, stream>>>(edge_src, edge_dst,
                                                        edge_val, PA, PB);
    tanh_rezero_kernel<<<(n4 + 255) / 256, 256, 0, stream>>>((float4*)PA,
                                                             (float4*)PB, n4);
  }
  // final softmax -> d_out
  softmax_out_kernel<<<(N_NODES * 32 + 255) / 256, 256, 0, stream>>>(
      PA, (float*)d_out);
}
